// IntraDomainScatterLoss_50182397887242
// MI455X (gfx1250) — matrix operand order repaired and checked by reading, not compiled, not run on hardware
//
#include <hip/hip_runtime.h>
#include <stdint.h>

// ---------------------------------------------------------------------------
// IntraDomainScatterLoss on MI455X (gfx1250, wave32, WMMA)
//
// loss = mean_d [ sum_{i,j in d} relu(1 - ||f_i - f_j||) / n_d^2 ]  over n_d>1
//
// Gram = F F^T (4096x4096x1024) via split-bf16 (hi+lo) WMMA 16x16x32 with f32
// accumulation; margin + per-domain masked reduction fused into the epilogue.
// Symmetric: only lower-triangular 128x128 blocks, off-diag weighted 2x.
//
// Fast path: F is split ONCE into bf16 hi/lo planes in workspace (L2-resident),
// then the GEMM stages tiles with GLOBAL_LOAD_ASYNC_TO_LDS_B128 (ASYNCcnt),
// double-buffered so the global fetch for tile kb+1 hides under tile kb's
// 192 WMMAs. Fallback (small workspace): split on the fly inside the GEMM.
// ---------------------------------------------------------------------------

typedef __attribute__((ext_vector_type(16))) __bf16 v16bf;
typedef __attribute__((ext_vector_type(8)))  float  v8f;

#define N_DOM 8
#define BM    128          // block tile (rows == cols)
#define BK    32           // K chunk (one WMMA K-step)
#define LDT   40           // padded LDS row stride in ushort (80 B: 16B-aligned rows)

__device__ __forceinline__ uint32_t f_as_u(float f)    { union { float f; uint32_t u; } c; c.f = f; return c.u; }
__device__ __forceinline__ float    u_as_f(uint32_t u) { union { float f; uint32_t u; } c; c.u = u; return c.f; }

union Frag { v16bf v; uint32_t u[8]; };

// Generic (flat) LDS pointer -> raw LDS byte offset (aperture low 32 bits).
__device__ __forceinline__ uint32_t lds_off(const void* p) {
    return (uint32_t)(uintptr_t)p;
}

// Async memory -> LDS copy, 16 bytes, GV addressing (64-bit VGPR address).
// Tracked by ASYNCcnt; completion via s_wait_asynccnt.
__device__ __forceinline__ void async_load_b128(uint32_t lds_byte_off, const void* gaddr) {
    asm volatile("global_load_async_to_lds_b128 %0, %1, off"
                 :: "v"(lds_byte_off), "v"(gaddr)
                 : "memory");
}
__device__ __forceinline__ void wait_async0() {
    asm volatile("s_wait_asynccnt 0x0" ::: "memory");
}

// --------------------------------------------- one-pass split + row norms
// Block r handles row r: 256 threads x 1 float4 = 1024 elements.
__global__ __launch_bounds__(256) void split_rowsq_kernel(const float* __restrict__ F,
                                                          unsigned short* __restrict__ Fhi,
                                                          unsigned short* __restrict__ Flo,
                                                          float* __restrict__ sq, int D) {
    const int r   = blockIdx.x;
    const int tid = threadIdx.x;
    float s = 0.f;
    for (int c4 = tid * 4; c4 < D; c4 += 256 * 4) {
        const float4 v = *(const float4*)(F + (size_t)r * D + c4);
        const float f[4] = {v.x, v.y, v.z, v.w};
        unsigned short h[4], l[4];
#pragma unroll
        for (int q = 0; q < 4; ++q) {
            s += f[q] * f[q];
            h[q] = (unsigned short)(f_as_u(f[q]) >> 16);
            const float lf = f[q] - u_as_f((uint32_t)h[q] << 16);
            l[q] = (unsigned short)(f_as_u(lf) >> 16);
        }
        const uint32_t hi01 = (uint32_t)h[0] | ((uint32_t)h[1] << 16);
        const uint32_t hi23 = (uint32_t)h[2] | ((uint32_t)h[3] << 16);
        const uint32_t lo01 = (uint32_t)l[0] | ((uint32_t)l[1] << 16);
        const uint32_t lo23 = (uint32_t)l[2] | ((uint32_t)l[3] << 16);
        *(uint2*)(Fhi + (size_t)r * D + c4) = make_uint2(hi01, hi23);
        *(uint2*)(Flo + (size_t)r * D + c4) = make_uint2(lo01, lo23);
    }
    __shared__ float red[256];
    red[tid] = s;
    __syncthreads();
    for (int off = 128; off > 0; off >>= 1) {
        if (tid < off) red[tid] += red[tid + off];
        __syncthreads();
    }
    if (tid == 0) sq[r] = red[0];
}

// ------------------------------------- plain row norms (fallback path only)
__global__ __launch_bounds__(256) void rowsq_kernel(const float* __restrict__ F,
                                                    float* __restrict__ sq, int D) {
    const int r = blockIdx.x;
    const float* p = F + (size_t)r * D;
    float s = 0.f;
    for (int k = threadIdx.x; k < D; k += 256) { float v = p[k]; s += v * v; }
    __shared__ float red[256];
    red[threadIdx.x] = s;
    __syncthreads();
    for (int off = 128; off > 0; off >>= 1) {
        if (threadIdx.x < off) red[threadIdx.x] += red[threadIdx.x + off];
        __syncthreads();
    }
    if (threadIdx.x == 0) sq[r] = red[0];
}

// ---------------------------------------------------------- shared epilogue
__device__ __forceinline__ void gram_epilogue(const v8f acc[4][2],
                                              const float* __restrict__ sq,
                                              const int* rowLab, const int* colLab,
                                              float* dsum, float* __restrict__ gdom,
                                              int rbase, int cbase, float w,
                                              int wm, int wn, int half, int lm, int tid) {
#pragma unroll
    for (int m = 0; m < 4; ++m) {
#pragma unroll
        for (int n = 0; n < 2; ++n) {
#pragma unroll
            for (int v = 0; v < 8; ++v) {
                const int li_loc = wm * 64 + m * 16 + v + half * 8;  // C layout: M = v + 8*half
                const int lj_loc = wn * 32 + n * 16 + lm;            // N = lane & 15
                const int gi = rbase + li_loc, gj = cbase + lj_loc;
                float d2 = sq[gi] + sq[gj] - 2.f * acc[m][n][v];
                if (gi == gj) d2 = 0.f;                 // exact-zero diagonal
                d2 = fmaxf(d2, 0.f);
                const float dist = (d2 > 0.f) ? sqrtf(d2) : 0.f;
                const float val  = fmaxf(1.f - dist, 0.f);
                const int li = rowLab[li_loc], lj = colLab[lj_loc];
                if (li == lj && val != 0.f) atomicAdd(&dsum[li], w * val);
            }
        }
    }
    __syncthreads();
    if (tid < N_DOM && dsum[tid] != 0.f) atomicAdd(&gdom[tid], dsum[tid]);
}

// ---------------- fast path: pre-split bf16, async double-buffered staging
__global__ __launch_bounds__(256) void gram_loss_async_kernel(
        const unsigned short* __restrict__ Fhi,
        const unsigned short* __restrict__ Flo,
        const int*   __restrict__ lab,
        const float* __restrict__ sq,
        float*       __restrict__ gdom,
        int N, int D) {
    __shared__ unsigned short Ahi[2][BM][LDT], Alo[2][BM][LDT];
    __shared__ unsigned short Bhi[2][BM][LDT], Blo[2][BM][LDT];
    __shared__ int   rowLab[BM], colLab[BM];
    __shared__ float dsum[N_DOM];

    int t = blockIdx.x, bi = 0;
    while (t > bi) { t -= bi + 1; ++bi; }
    const int bj    = t;
    const int rbase = bi * BM, cbase = bj * BM;
    const float w   = (bi == bj) ? 1.f : 2.f;

    const int tid = threadIdx.x;
    if (tid < N_DOM) dsum[tid] = 0.f;
    for (int i = tid; i < BM; i += 256) {
        rowLab[i] = lab[rbase + i];
        colLab[i] = lab[cbase + i];
    }

    const int wave = tid >> 5, lane = tid & 31;
    const int wm = wave & 1, wn = wave >> 1;
    const int half = lane >> 4, lm = lane & 15;

    v8f acc[4][2];
#pragma unroll
    for (int m = 0; m < 4; ++m)
#pragma unroll
        for (int n = 0; n < 2; ++n)
            acc[m][n] = (v8f){0.f, 0.f, 0.f, 0.f, 0.f, 0.f, 0.f, 0.f};

    // Staging: per array 128 rows x 32 ushorts = 512 x 16B slots, 2 per thread.
    const int r0 = tid >> 2,          c80 = (tid & 3) << 3;
    const int r1 = (tid + 256) >> 2,  c81 = c80;          // second slot, rows 64..127
    const size_t gA0 = (size_t)(rbase + r0) * D + c80;
    const size_t gA1 = (size_t)(rbase + r1) * D + c81;
    const size_t gB0 = (size_t)(cbase + r0) * D + c80;
    const size_t gB1 = (size_t)(cbase + r1) * D + c81;

    const int KB = D / BK;

    // Prologue: issue tile 0 into buffer 0.
    {
        async_load_b128(lds_off(&Ahi[0][r0][c80]), Fhi + gA0);
        async_load_b128(lds_off(&Ahi[0][r1][c81]), Fhi + gA1);
        async_load_b128(lds_off(&Alo[0][r0][c80]), Flo + gA0);
        async_load_b128(lds_off(&Alo[0][r1][c81]), Flo + gA1);
        async_load_b128(lds_off(&Bhi[0][r0][c80]), Fhi + gB0);
        async_load_b128(lds_off(&Bhi[0][r1][c81]), Fhi + gB1);
        async_load_b128(lds_off(&Blo[0][r0][c80]), Flo + gB0);
        async_load_b128(lds_off(&Blo[0][r1][c81]), Flo + gB1);
    }

    for (int kb = 0; kb < KB; ++kb) {
        const int p = kb & 1;
        wait_async0();        // our async writes for buffer p have landed
        __syncthreads();      // everyone's writes visible; buffer p^1 free

        if (kb + 1 < KB) {    // prefetch tile kb+1 into buffer p^1 (hidden)
            const int q = p ^ 1;
            const size_t ko = (size_t)(kb + 1) * BK;
            async_load_b128(lds_off(&Ahi[q][r0][c80]), Fhi + gA0 + ko);
            async_load_b128(lds_off(&Ahi[q][r1][c81]), Fhi + gA1 + ko);
            async_load_b128(lds_off(&Alo[q][r0][c80]), Flo + gA0 + ko);
            async_load_b128(lds_off(&Alo[q][r1][c81]), Flo + gA1 + ko);
            async_load_b128(lds_off(&Bhi[q][r0][c80]), Fhi + gB0 + ko);
            async_load_b128(lds_off(&Bhi[q][r1][c81]), Fhi + gB1 + ko);
            async_load_b128(lds_off(&Blo[q][r0][c80]), Flo + gB0 + ko);
            async_load_b128(lds_off(&Blo[q][r1][c81]), Flo + gB1 + ko);
        }

        Frag ah[4], al[4], bh[2], bl[2];
#pragma unroll
        for (int m = 0; m < 4; ++m) {
            const int row = wm * 64 + m * 16 + lm;
            const unsigned short* ph = &Ahi[p][row][0];
            const unsigned short* pl = &Alo[p][row][0];
#pragma unroll
            for (int j = 0; j < 8; ++j) {
                const int k = (j < 4 ? 2 * j : 16 + 2 * (j - 4)) + half * 8;
                ah[m].u[j] = *(const uint32_t*)(ph + k);
                al[m].u[j] = *(const uint32_t*)(pl + k);
            }
        }
#pragma unroll
        for (int n = 0; n < 2; ++n) {
            const int row = wn * 32 + n * 16 + lm;
            const unsigned short* ph = &Bhi[p][row][0];
            const unsigned short* pl = &Blo[p][row][0];
#pragma unroll
            for (int j = 0; j < 8; ++j) {
                const int k = (j < 4 ? 2 * j : 16 + 2 * (j - 4)) + half * 8;
                bh[n].u[j] = *(const uint32_t*)(ph + k);
                bl[n].u[j] = *(const uint32_t*)(pl + k);
            }
        }

#pragma unroll
        for (int m = 0; m < 4; ++m)
#pragma unroll
            for (int n = 0; n < 2; ++n) {
                acc[m][n] = __builtin_amdgcn_wmma_f32_16x16x32_bf16(
                    false, ah[m].v, false, bh[n].v, (short)0, acc[m][n], false, false);
                acc[m][n] = __builtin_amdgcn_wmma_f32_16x16x32_bf16(
                    false, ah[m].v, false, bl[n].v, (short)0, acc[m][n], false, false);
                acc[m][n] = __builtin_amdgcn_wmma_f32_16x16x32_bf16(
                    false, al[m].v, false, bh[n].v, (short)0, acc[m][n], false, false);
            }
    }
    __syncthreads();
    gram_epilogue(acc, sq, rowLab, colLab, dsum, gdom, rbase, cbase, w, wm, wn, half, lm, tid);
}

// ---------------------------- fallback path: split fp32 -> bf16 per K-step
__global__ __launch_bounds__(256) void gram_loss_fallback_kernel(
        const float* __restrict__ F,
        const int*   __restrict__ lab,
        const float* __restrict__ sq,
        float*       __restrict__ gdom,
        int N, int D) {
    __shared__ unsigned short Ahi[BM][LDT], Alo[BM][LDT];
    __shared__ unsigned short Bhi[BM][LDT], Blo[BM][LDT];
    __shared__ int   rowLab[BM], colLab[BM];
    __shared__ float dsum[N_DOM];

    int t = blockIdx.x, bi = 0;
    while (t > bi) { t -= bi + 1; ++bi; }
    const int bj    = t;
    const int rbase = bi * BM, cbase = bj * BM;
    const float w   = (bi == bj) ? 1.f : 2.f;

    const int tid = threadIdx.x;
    if (tid < N_DOM) dsum[tid] = 0.f;
    for (int i = tid; i < BM; i += 256) {
        rowLab[i] = lab[rbase + i];
        colLab[i] = lab[cbase + i];
    }

    const int wave = tid >> 5, lane = tid & 31;
    const int wm = wave & 1, wn = wave >> 1;
    const int half = lane >> 4, lm = lane & 15;

    v8f acc[4][2];
#pragma unroll
    for (int m = 0; m < 4; ++m)
#pragma unroll
        for (int n = 0; n < 2; ++n)
            acc[m][n] = (v8f){0.f, 0.f, 0.f, 0.f, 0.f, 0.f, 0.f, 0.f};

    const int KB = D / BK;
    for (int kb = 0; kb < KB; ++kb) {
        __syncthreads();
#pragma unroll
        for (int it = 0; it < 4; ++it) {
            const int idx = tid + it * 256;      // 0..1023 float4 slots
            const int r   = idx >> 3;
            const int c4  = (idx & 7) << 2;
            const float4 va = *(const float4*)(F + (size_t)(rbase + r) * D + kb * BK + c4);
            const float4 vb = *(const float4*)(F + (size_t)(cbase + r) * D + kb * BK + c4);
            const float fa[4] = {va.x, va.y, va.z, va.w};
            const float fb[4] = {vb.x, vb.y, vb.z, vb.w};
#pragma unroll
            for (int q = 0; q < 4; ++q) {
                unsigned short h = (unsigned short)(f_as_u(fa[q]) >> 16);
                float lf = fa[q] - u_as_f((uint32_t)h << 16);
                Ahi[r][c4 + q] = h;
                Alo[r][c4 + q] = (unsigned short)(f_as_u(lf) >> 16);
                unsigned short hb = (unsigned short)(f_as_u(fb[q]) >> 16);
                float lb = fb[q] - u_as_f((uint32_t)hb << 16);
                Bhi[r][c4 + q] = hb;
                Blo[r][c4 + q] = (unsigned short)(f_as_u(lb) >> 16);
            }
        }
        __syncthreads();

        Frag ah[4], al[4], bh[2], bl[2];
#pragma unroll
        for (int m = 0; m < 4; ++m) {
            const int row = wm * 64 + m * 16 + lm;
            const unsigned short* ph = &Ahi[row][0];
            const unsigned short* pl = &Alo[row][0];
#pragma unroll
            for (int j = 0; j < 8; ++j) {
                const int k = (j < 4 ? 2 * j : 16 + 2 * (j - 4)) + half * 8;
                ah[m].u[j] = *(const uint32_t*)(ph + k);
                al[m].u[j] = *(const uint32_t*)(pl + k);
            }
        }
#pragma unroll
        for (int n = 0; n < 2; ++n) {
            const int row = wn * 32 + n * 16 + lm;
            const unsigned short* ph = &Bhi[row][0];
            const unsigned short* pl = &Blo[row][0];
#pragma unroll
            for (int j = 0; j < 8; ++j) {
                const int k = (j < 4 ? 2 * j : 16 + 2 * (j - 4)) + half * 8;
                bh[n].u[j] = *(const uint32_t*)(ph + k);
                bl[n].u[j] = *(const uint32_t*)(pl + k);
            }
        }

#pragma unroll
        for (int m = 0; m < 4; ++m)
#pragma unroll
            for (int n = 0; n < 2; ++n) {
                acc[m][n] = __builtin_amdgcn_wmma_f32_16x16x32_bf16(
                    false, ah[m].v, false, bh[n].v, (short)0, acc[m][n], false, false);
                acc[m][n] = __builtin_amdgcn_wmma_f32_16x16x32_bf16(
                    false, ah[m].v, false, bl[n].v, (short)0, acc[m][n], false, false);
                acc[m][n] = __builtin_amdgcn_wmma_f32_16x16x32_bf16(
                    false, al[m].v, false, bh[n].v, (short)0, acc[m][n], false, false);
            }
    }
    __syncthreads();
    gram_epilogue(acc, sq, rowLab, colLab, dsum, gdom, rbase, cbase, w, wm, wn, half, lm, tid);
}

// ----------------------------------------------------------------- finalize
__global__ __launch_bounds__(256) void finalize_kernel(const int* __restrict__ lab,
                                                       const float* __restrict__ gdom,
                                                       float* __restrict__ out, int N) {
    __shared__ float cnt[N_DOM];
    if (threadIdx.x < N_DOM) cnt[threadIdx.x] = 0.f;
    __syncthreads();
    for (int i = threadIdx.x; i < N; i += 256) atomicAdd(&cnt[lab[i]], 1.f);
    __syncthreads();
    if (threadIdx.x == 0) {
        float s = 0.f, c = 0.f;
        for (int d = 0; d < N_DOM; ++d) {
            const float n = cnt[d];
            if (n > 1.f) { s += gdom[d] / (n * n); c += 1.f; }
        }
        out[0] = (c > 0.f) ? s / c : 0.f;
    }
}

// ------------------------------------------------------------------- launch
extern "C" void kernel_launch(void* const* d_in, const int* in_sizes, int n_in,
                              void* d_out, int out_size, void* d_ws, size_t ws_size,
                              hipStream_t stream) {
    const float* F   = (const float*)d_in[0];
    const int*   lab = (const int*)d_in[1];
    const int N = in_sizes[1];
    const int D = in_sizes[0] / N;
    const int nb   = N / BM;
    const int grid = nb * (nb + 1) / 2;            // lower-triangular blocks

    // Workspace layout: [sq: N floats][gdom: 8 floats][pad][Fhi][Flo]
    float* sq   = (float*)d_ws;
    float* gdom = sq + N;
    size_t off  = ((size_t)N * sizeof(float) + N_DOM * sizeof(float) + 255) & ~(size_t)255;
    unsigned short* Fhi = (unsigned short*)((char*)d_ws + off);
    unsigned short* Flo = Fhi + (size_t)N * D;
    const size_t need = off + 2 * (size_t)N * D * sizeof(unsigned short);

    hipMemsetAsync(gdom, 0, N_DOM * sizeof(float), stream);

    if (ws_size >= need) {
        split_rowsq_kernel<<<N, 256, 0, stream>>>(F, Fhi, Flo, sq, D);
        gram_loss_async_kernel<<<grid, 256, 0, stream>>>(Fhi, Flo, lab, sq, gdom, N, D);
    } else {
        rowsq_kernel<<<N, 256, 0, stream>>>(F, sq, D);
        gram_loss_fallback_kernel<<<grid, 256, 0, stream>>>(F, lab, sq, gdom, N, D);
    }
    finalize_kernel<<<1, 256, 0, stream>>>(lab, gdom, (float*)d_out, N);
}
